// PhysNetModule_55035710931189
// MI455X (gfx1250) — matrix operand order repaired and checked by reading, not compiled, not run on hardware
//
#include <hip/hip_runtime.h>

#define D 128
#define WT_STRIDE 132   // padded LDS row stride (halves) for 128-wide tiles (bank-conflict free)
#define AG_STRIDE 36    // padded LDS row stride (halves) for 32-wide tiles

typedef __attribute__((ext_vector_type(16))) _Float16 v16h;
typedef __attribute__((ext_vector_type(8)))  float    v8f;

// shifted softplus via raw v_exp_f32/v_log_f32 (base-2), branch-free:
// ssp(x) = max(x,0) + ln2*log2(1 + 2^(-|x|*log2e)) - ln2 ; arg of log2 is in (1,2], never denormal
__device__ __forceinline__ float sspf(float x) {
    const float LOG2E = 1.4426950408889634f;
    const float LN2   = 0.6931471805599453f;
    float t = __builtin_amdgcn_exp2f(-fabsf(x) * LOG2E);   // e^{-|x|}
    float l = __builtin_amdgcn_logf(1.0f + t);             // log2(1+e^{-|x|})
    return fmaxf(x, 0.0f) + (l * LN2 - LN2);
}

__device__ __forceinline__ v8f wmma16(v16h a, v16h b, v8f c) {
    return __builtin_amdgcn_wmma_f32_16x16x32_f16(false, a, false, b, (short)0, c, false, false);
}

// A fragment (16xK tile, f16): lanes 0-15 M=lane; halves t -> K=kbase+16g+t contiguous
__device__ __forceinline__ v16h load_a_frag(const _Float16* tile, int lane, int kbase, int stride) {
    const int M = lane & 15, g = lane >> 4;
    const _Float16* p0 = tile + M * stride + kbase + 16 * g;
    v16h r;
#pragma unroll
    for (int i = 0; i < 8; ++i) { r[i] = p0[i]; r[8 + i] = p0[8 + i]; }
    return r;
}

// B fragment from TRANSPOSED weight tile wt[n][k]: lane holds col N=lane&15, K=kbase+16g+t contiguous
__device__ __forceinline__ v16h load_b_frag(const _Float16* wt, int lane, int ntile, int kbase, int stride) {
    const int n = lane & 15, g = lane >> 4;
    const _Float16* p = wt + (ntile * 16 + n) * stride + kbase + 16 * g;
    v16h r;
#pragma unroll
    for (int i = 0; i < 16; ++i) r[i] = p[i];
    return r;
}

__device__ __forceinline__ void load_weight_tr(const float* __restrict__ W, _Float16* wt,
                                               int tid, int nthreads) {
    for (int i = tid; i < D * D; i += nthreads) {
        int k = i >> 7, n = i & 127;          // W[k][n] row-major
        wt[n * WT_STRIDE + k] = (_Float16)W[i];
    }
}

// ---------------- prep: xa_h = f16(ssp(x)), xp = u * ssp(x) ----------------
__global__ void prep_kernel(const float* __restrict__ x, const float* __restrict__ u,
                            _Float16* __restrict__ xa_h, float* __restrict__ xp, int total) {
    int i = blockIdx.x * blockDim.x + threadIdx.x;
    if (i < total) {
        float a = sspf(x[i]);
        xa_h[i] = (_Float16)a;
        xp[i]   = u[i & (D - 1)] * a;
    }
}

// ---------------- generic single GEMM: Y = [ssp?]( [ssp?](X) @ W + b (+ addend) ) ----------------
template <int PRE, int POST, int HASADD>
__global__ void __launch_bounds__(128)
gemm_kernel(const float* __restrict__ X, const float* __restrict__ W, const float* __restrict__ bias,
            const float* __restrict__ addend, float* __restrict__ Y, int nrows) {
    __shared__ _Float16 wt[D * WT_STRIDE];
    __shared__ _Float16 at[4][16 * WT_STRIDE];
    const int tid = threadIdx.x, lane = tid & 31, wave = tid >> 5;
    load_weight_tr(W, wt, tid, blockDim.x);
    const int Ncol = lane & 15, Mbase = 8 * (lane >> 4);
    const int off0 = Mbase * D + Ncol;          // 32-bit per-lane offset, base stays scalar
    float breg[8];
#pragma unroll
    for (int nt = 0; nt < 8; ++nt) breg[nt] = bias[nt * 16 + Ncol];
    __syncthreads();
    _Float16* A = at[wave];
    const int ntiles = (nrows + 15) >> 4;
    for (int tile = blockIdx.x * 4 + wave; tile < ntiles; tile += gridDim.x * 4) {
        const int row0 = tile * 16;
        // build A tile: per it one (uniform) row, lanes cover 4 floats each
#pragma unroll
        for (int it = 0; it < 16; ++it) {
            int row = row0 + it; if (row >= nrows) row = nrows - 1;
            const float4 f = *reinterpret_cast<const float4*>(X + (size_t)row * D + lane * 4);
            float v0 = f.x, v1 = f.y, v2 = f.z, v3 = f.w;
            if (PRE) { v0 = sspf(v0); v1 = sspf(v1); v2 = sspf(v2); v3 = sspf(v3); }
            _Float16* p = A + it * WT_STRIDE + lane * 4;
            p[0] = (_Float16)v0; p[1] = (_Float16)v1; p[2] = (_Float16)v2; p[3] = (_Float16)v3;
        }
        v16h af[4];
#pragma unroll
        for (int kt = 0; kt < 4; ++kt) af[kt] = load_a_frag(A, lane, kt * 32, WT_STRIDE);
        v8f acc[8];
#pragma unroll
        for (int nt = 0; nt < 8; ++nt) acc[nt] = (v8f){};
        // kt-outer: 8 independent accumulator chains -> back-to-back WMMAs without RAW stalls
#pragma unroll
        for (int kt = 0; kt < 4; ++kt)
#pragma unroll
            for (int nt = 0; nt < 8; ++nt)
                acc[nt] = wmma16(af[kt], load_b_frag(wt, lane, nt, kt * 32, WT_STRIDE), acc[nt]);
        float* yb = Y + (size_t)row0 * D;
        const float* ab = addend + (size_t)row0 * D;    // only dereferenced when HASADD
        if (row0 + 16 <= nrows) {                       // full tile: branchless epilogue
#pragma unroll
            for (int nt = 0; nt < 8; ++nt) {
#pragma unroll
                for (int r = 0; r < 8; ++r) {
                    float vv = acc[nt][r] + breg[nt];
                    if (HASADD) vv += ab[off0 + r * D + nt * 16];
                    if (POST) vv = sspf(vv);
                    yb[off0 + r * D + nt * 16] = vv;
                }
            }
        } else {
#pragma unroll
            for (int nt = 0; nt < 8; ++nt) {
#pragma unroll
                for (int r = 0; r < 8; ++r) {
                    if (row0 + Mbase + r < nrows) {
                        float vv = acc[nt][r] + breg[nt];
                        if (HASADD) vv += ab[off0 + r * D + nt * 16];
                        if (POST) vv = sspf(vv);
                        yb[off0 + r * D + nt * 16] = vv;
                    }
                }
            }
        }
    }
}

// ---------------- fused residual: Y = [ssp?]( ssp(ssp(X)@W+b)@W + b + X ) ----------------
template <int POST>
__global__ void __launch_bounds__(128)
residual_kernel(const float* __restrict__ X, const float* __restrict__ W, const float* __restrict__ bias,
                float* __restrict__ Y, int nrows) {
    __shared__ _Float16 wt[D * WT_STRIDE];
    __shared__ _Float16 at[4][16 * WT_STRIDE];
    const int tid = threadIdx.x, lane = tid & 31, wave = tid >> 5;
    load_weight_tr(W, wt, tid, blockDim.x);
    const int Ncol = lane & 15, Mbase = 8 * (lane >> 4);
    const int off0 = Mbase * D + Ncol;
    float breg[8];
#pragma unroll
    for (int nt = 0; nt < 8; ++nt) breg[nt] = bias[nt * 16 + Ncol];
    __syncthreads();
    _Float16* A = at[wave];
    const int ntiles = (nrows + 15) >> 4;
    for (int tile = blockIdx.x * 4 + wave; tile < ntiles; tile += gridDim.x * 4) {
        const int row0 = tile * 16;
#pragma unroll
        for (int it = 0; it < 16; ++it) {
            int row = row0 + it; if (row >= nrows) row = nrows - 1;
            const float4 f = *reinterpret_cast<const float4*>(X + (size_t)row * D + lane * 4);
            _Float16* p = A + it * WT_STRIDE + lane * 4;
            p[0] = (_Float16)sspf(f.x); p[1] = (_Float16)sspf(f.y);
            p[2] = (_Float16)sspf(f.z); p[3] = (_Float16)sspf(f.w);
        }
        v16h af[4];
#pragma unroll
        for (int kt = 0; kt < 4; ++kt) af[kt] = load_a_frag(A, lane, kt * 32, WT_STRIDE);
        v8f acc[8];
#pragma unroll
        for (int nt = 0; nt < 8; ++nt) acc[nt] = (v8f){};
#pragma unroll
        for (int kt = 0; kt < 4; ++kt)
#pragma unroll
            for (int nt = 0; nt < 8; ++nt)
                acc[nt] = wmma16(af[kt], load_b_frag(wt, lane, nt, kt * 32, WT_STRIDE), acc[nt]);
        // overwrite A tile with ssp(GEMM1 + b) for the second GEMM (same W)
        _Float16* Ab = A + Mbase * WT_STRIDE + Ncol;
#pragma unroll
        for (int nt = 0; nt < 8; ++nt)
#pragma unroll
            for (int r = 0; r < 8; ++r)
                Ab[r * WT_STRIDE + nt * 16] = (_Float16)sspf(acc[nt][r] + breg[nt]);
#pragma unroll
        for (int kt = 0; kt < 4; ++kt) af[kt] = load_a_frag(A, lane, kt * 32, WT_STRIDE);
#pragma unroll
        for (int nt = 0; nt < 8; ++nt) acc[nt] = (v8f){};
#pragma unroll
        for (int kt = 0; kt < 4; ++kt)
#pragma unroll
            for (int nt = 0; nt < 8; ++nt)
                acc[nt] = wmma16(af[kt], load_b_frag(wt, lane, nt, kt * 32, WT_STRIDE), acc[nt]);
        float* yb = Y + (size_t)row0 * D;
        const float* xb = X + (size_t)row0 * D;
        if (row0 + 16 <= nrows) {
#pragma unroll
            for (int nt = 0; nt < 8; ++nt) {
#pragma unroll
                for (int r = 0; r < 8; ++r) {
                    float vv = acc[nt][r] + breg[nt] + xb[off0 + r * D + nt * 16];
                    if (POST) vv = sspf(vv);
                    yb[off0 + r * D + nt * 16] = vv;
                }
            }
        } else {
#pragma unroll
            for (int nt = 0; nt < 8; ++nt) {
#pragma unroll
                for (int r = 0; r < 8; ++r) {
                    if (row0 + Mbase + r < nrows) {
                        float vv = acc[nt][r] + breg[nt] + xb[off0 + r * D + nt * 16];
                        if (POST) vv = sspf(vv);
                        yb[off0 + r * D + nt * 16] = vv;
                    }
                }
            }
        }
    }
}

// ---------------- edge kernel: v[idx_i] += ssp(xa[idx_j]@Wj + bj) * (g@Wg) ----------------
__global__ void __launch_bounds__(128)
edge_kernel(const _Float16* __restrict__ xa_h, const float* __restrict__ g,
            const float* __restrict__ Wj, const float* __restrict__ bj,
            const float* __restrict__ Wg, const int* __restrict__ idx_i,
            const int* __restrict__ idx_j, float* __restrict__ v, int E) {
    __shared__ _Float16 wjt[D * WT_STRIDE];
    __shared__ _Float16 wgt[D * AG_STRIDE];
    __shared__ _Float16 axa[4][16 * WT_STRIDE];
    __shared__ _Float16 agt[4][16 * AG_STRIDE];
    const int tid = threadIdx.x, lane = tid & 31, wave = tid >> 5;
    load_weight_tr(Wj, wjt, tid, blockDim.x);
    for (int i = tid; i < 32 * D; i += blockDim.x) {
        int k = i >> 7, n = i & 127;                  // Wg[k][n], k<32
        wgt[n * AG_STRIDE + k] = (_Float16)Wg[i];
    }
    const int Ncol = lane & 15, gg = lane >> 4;
    float breg[8];
#pragma unroll
    for (int nt = 0; nt < 8; ++nt) breg[nt] = bj[nt * 16 + Ncol];
    __syncthreads();
    _Float16* A = axa[wave];
    _Float16* G = agt[wave];
    const int ntiles = (E + 15) >> 4;
    const int tstep = gridDim.x * 4;
    for (int tile = blockIdx.x * 4 + wave; tile < ntiles; tile += tstep) {
        const int e0 = tile * 16;
        // prefetch next tile's index/RBF data (global_prefetch_b8)
        {
            int en = (tile + tstep) * 16; if (en >= E) en = 0;
            __builtin_prefetch(idx_j + en, 0, 1);
            __builtin_prefetch(idx_i + en, 0, 1);
            __builtin_prefetch(g + (size_t)en * 32, 0, 1);
        }
        // gather 16 xa rows via idx_j: per it the row (and idx_j) is wave-uniform -> scalar base
#pragma unroll
        for (int it = 0; it < 16; ++it) {
            int e = e0 + it; if (e >= E) e = E - 1;
            const int j = idx_j[e];
            const float2 f = *reinterpret_cast<const float2*>(xa_h + (size_t)j * D + lane * 4);
            *reinterpret_cast<float2*>(A + it * WT_STRIDE + lane * 4) = f;
        }
        // 16x32 RBF tile; zero out-of-range rows so cg==0 masks tail edges (branchless epilogue)
#pragma unroll
        for (int it = 0; it < 16; ++it) {
            const int e = e0 + it;
            const int ec = (e < E) ? e : (E - 1);
            float gv = g[(size_t)ec * 32 + lane];
            if (e >= E) gv = 0.0f;
            G[it * AG_STRIDE + lane] = (_Float16)gv;
        }
        // per-lane scatter offsets (32-bit; base pointer stays scalar -> GVS addressing)
        int voff[8];
#pragma unroll
        for (int r = 0; r < 8; ++r) {
            const int e = e0 + 8 * gg + r;
            const int ir = (e < E) ? idx_i[e] : 0;   // tail adds 0.0 to row 0
            voff[r] = ir * D + Ncol;
        }
        v16h af[4];
#pragma unroll
        for (int kt = 0; kt < 4; ++kt) af[kt] = load_a_frag(A, lane, kt * 32, WT_STRIDE);
        const v16h agf = load_a_frag(G, lane, 0, AG_STRIDE);
        v8f acc[8];
#pragma unroll
        for (int nt = 0; nt < 8; ++nt) acc[nt] = (v8f){};
#pragma unroll
        for (int kt = 0; kt < 4; ++kt)
#pragma unroll
            for (int nt = 0; nt < 8; ++nt)
                acc[nt] = wmma16(af[kt], load_b_frag(wjt, lane, nt, kt * 32, WT_STRIDE), acc[nt]);
#pragma unroll
        for (int nt = 0; nt < 8; ++nt) {
            v8f cg = {};
            cg = wmma16(agf, load_b_frag(wgt, lane, nt, 0, AG_STRIDE), cg);
#pragma unroll
            for (int r = 0; r < 8; ++r) {
                const float val = sspf(acc[nt][r] + breg[nt]) * cg[r];
                atomicAdd(&v[voff[r] + nt * 16], val);
            }
        }
    }
}

extern "C" void kernel_launch(void* const* d_in, const int* in_sizes, int n_in,
                              void* d_out, int out_size, void* d_ws, size_t ws_size,
                              hipStream_t stream) {
    const float* x      = (const float*)d_in[0];
    const float* g      = (const float*)d_in[1];
    const float* Wf     = (const float*)d_in[2];
    const float* bf     = (const float*)d_in[3];
    const float* Wg     = (const float*)d_in[4];
    const float* Wj     = (const float*)d_in[5];
    const float* bj     = (const float*)d_in[6];
    const float* Wi     = (const float*)d_in[7];
    const float* bi     = (const float*)d_in[8];
    const float* u      = (const float*)d_in[9];
    const float* rWint  = (const float*)d_in[10];
    const float* rbint  = (const float*)d_in[11];
    const float* rWatom = (const float*)d_in[12];
    const float* rbatom = (const float*)d_in[13];
    const float* rWout  = (const float*)d_in[14];
    const float* rbout  = (const float*)d_in[15];
    const int*   idx_i  = (const int*)d_in[16];
    const int*   idx_j  = (const int*)d_in[17];

    const int N = in_sizes[0] / D;
    const int E = in_sizes[16];
    const int n_int  = in_sizes[10] / (D * D);
    const int n_atom = in_sizes[12] / (D * D);
    const int n_outr = in_sizes[14] / (D * D);

    const size_t nd = (size_t)N * D;
    float*    xp   = (float*)d_ws;
    float*    v    = xp + nd;
    float*    h    = v + nd;
    _Float16* xa_h = (_Float16*)(h + nd);

    const int total = (int)nd;
    prep_kernel<<<(total + 255) / 256, 256, 0, stream>>>(x, u, xa_h, xp, total);

    const int ntiles  = (N + 15) / 16;
    const int gblocks = (ntiles + 3) / 4;   // 4 waves/block, one 16-row tile per wave

    // v = vm = ssp(ssp(x)@Wi + bi)
    gemm_kernel<1, 1, 0><<<gblocks, 128, 0, stream>>>(x, Wi, bi, nullptr, v, N);
    // v += segment_sum(ssp(xa_j@Wj+bj) * (g@Wg), idx_i)
    edge_kernel<<<1024, 128, 0, stream>>>(xa_h, g, Wj, bj, Wg, idx_i, idx_j, v, E);
    // interaction residual stack
    for (int k = 0; k < n_int; ++k)
        residual_kernel<0><<<gblocks, 128, 0, stream>>>(v, rWint + (size_t)k * D * D,
                                                        rbint + (size_t)k * D, v, N);
    // h = xp + ssp(v)@Wf + bf
    gemm_kernel<1, 0, 1><<<gblocks, 128, 0, stream>>>(v, Wf, bf, xp, h, N);
    // atomic residual stack (in place on h)
    for (int k = 0; k < n_atom; ++k)
        residual_kernel<0><<<gblocks, 128, 0, stream>>>(h, rWatom + (size_t)k * D * D,
                                                        rbatom + (size_t)k * D, h, N);
    // output stack: o = ssp(residual(...)) -> d_out[0:N*D)
    float* o = (float*)d_out;
    for (int k = 0; k < n_outr; ++k) {
        const float* src = (k == 0) ? h : o;
        if (k == n_outr - 1)
            residual_kernel<1><<<gblocks, 128, 0, stream>>>(src, rWout + (size_t)k * D * D,
                                                            rbout + (size_t)k * D, o, N);
        else
            residual_kernel<0><<<gblocks, 128, 0, stream>>>(src, rWout + (size_t)k * D * D,
                                                            rbout + (size_t)k * D, o, N);
    }
    // second output: h -> d_out[N*D : 2*N*D)
    hipMemcpyAsync(o + nd, h, nd * sizeof(float), hipMemcpyDeviceToDevice, stream);
}
